// E3DeePH_34952443854882
// MI455X (gfx1250) — matrix operand order repaired and checked by reading, not compiled, not run on hardware
//
#include <hip/hip_runtime.h>
#include <math.h>

#define N_NODES 10000
#define N_EDGES 160000
#define C_CH    32
#define NBASIS  128
#define HDIM    64
#define RCUT    5.0f
#define PI_F    3.14159265358979323846f

typedef __bf16 bf16_t;
typedef bf16_t v16bf __attribute__((ext_vector_type(16)));
typedef float  v8f   __attribute__((ext_vector_type(8)));

__device__ __forceinline__ float siluf(float x)    { return x / (1.0f + __expf(-x)); }
__device__ __forceinline__ float sigmoidf_(float x){ return 1.0f / (1.0f + __expf(-x)); }

__device__ __forceinline__ void atomAddF(float* p, float v) {
  __hip_atomic_fetch_add(p, v, __ATOMIC_RELAXED, __HIP_MEMORY_SCOPE_AGENT);
}

// Build a 16x32 bf16 A-fragment (v16bf per lane) from an f32 LDS tile.
// ISA layout: lanes 0-15 -> rows m=lane, K = {0..7, 16..23}; lanes 16-31 -> K = {8..15, 24..31}.
__device__ __forceinline__ v16bf make_a_frag(const float* tile, int stride, int k0, int lane) {
  int m  = lane & 15;
  int kb = (lane < 16) ? 0 : 8;
  const float* row = tile + m * stride + k0;
  v16bf a;
#pragma unroll
  for (int i = 0; i < 8; ++i) a[i]     = (bf16_t)row[kb + i];
#pragma unroll
  for (int i = 0; i < 8; ++i) a[8 + i] = (bf16_t)row[kb + 16 + i];
  return a;
}

// Pre-swizzle an f32 KxN weight matrix into WMMA B-fragment order:
// buf[((kc*NT + nt)*32 + lane)*16 + i], k = kc*32 + (lane<16?0:8) + (i<8?i:i+8), n = nt*16 + (lane&15)
__global__ void prep_frag_kernel(const float* __restrict__ src, bf16_t* __restrict__ dst,
                                 int NT, int Ncols, int total) {
  int t = blockIdx.x * blockDim.x + threadIdx.x;
  if (t >= total) return;
  int i    = t & 15;
  int lane = (t >> 4) & 31;
  int rem  = t >> 9;
  int nt   = rem % NT;
  int kc   = rem / NT;
  int k = kc * 32 + ((lane < 16) ? 0 : 8) + ((i < 8) ? i : i + 8);
  int n = nt * 16 + (lane & 15);
  dst[t] = (bf16_t)src[k * Ncols + n];
}

// One wave handles 16 edges: radial basis -> MLP (WMMA bf16) -> w_all, ew; plus spherical harmonics.
__global__ __launch_bounds__(128)
void edge_mlp_kernel(const float* __restrict__ pos, const int* __restrict__ ei,
                     const float* __restrict__ rb1, const float* __restrict__ rb2,
                     const float* __restrict__ edge_b,
                     const v16bf* __restrict__ bfrag1, const v16bf* __restrict__ bfrag2,
                     const v16bf* __restrict__ bfragE,
                     float* __restrict__ sh_all, float* __restrict__ w_all,
                     float* __restrict__ ew_all) {
  __shared__ float rb_lds[4][16][NBASIS];
  __shared__ float h_lds[4][16][HDIM];
  __shared__ float geom[4][16][2];  // u, env*sqrt(2/RC)/r

  int wid  = threadIdx.x >> 5;
  int lane = threadIdx.x & 31;
  long base = ((long)blockIdx.x * 4 + wid) * 16;

  if (lane < 16) {
    long e = base + lane;
    if (e < N_EDGES) {
      int s = ei[e], d = ei[N_EDGES + e];
      float vx = pos[d * 3 + 0] - pos[s * 3 + 0];
      float vy = pos[d * 3 + 1] - pos[s * 3 + 1];
      float vz = pos[d * 3 + 2] - pos[s * 3 + 2];
      float r = sqrtf(vx * vx + vy * vy + vz * vz + 1e-12f);
      float u = fminf(fmaxf(r / RCUT, 0.0f), 1.0f);
      float env = 0.5f * (__cosf(PI_F * u) + 1.0f);
      geom[wid][lane][0] = u;
      geom[wid][lane][1] = env * sqrtf(2.0f / RCUT) / r;
      float x = vx / r, y = vy / r, z = vz / r;
      const float s3 = 1.7320508075688772f, s5 = 2.23606797749979f, s15 = 3.872983346207417f;
      const float c1 = 2.091650066335189f;   // sqrt(35/8)
      const float c2 = 10.246950765959598f;  // sqrt(105)
      const float c3 = 1.6201851746019651f;  // sqrt(21/8)
      const float c4 = 1.3228756555322954f;  // sqrt(7)/2
      const float c5 = 5.123475382979799f;   // sqrt(105)/2
      float* sh = &sh_all[e * 16];
      sh[0] = 1.0f;
      sh[1] = s3 * x; sh[2] = s3 * y; sh[3] = s3 * z;
      sh[4] = s15 * x * y;
      sh[5] = s15 * y * z;
      sh[6] = 0.5f * s5 * (3.0f * z * z - 1.0f);
      sh[7] = s15 * x * z;
      sh[8] = 0.5f * s15 * (x * x - y * y);
      sh[9]  = c1 * y * (3.0f * x * x - y * y);
      sh[10] = c2 * x * y * z;
      sh[11] = c3 * y * (5.0f * z * z - 1.0f);
      sh[12] = c4 * (5.0f * z * z * z - 3.0f * z);
      sh[13] = c3 * x * (5.0f * z * z - 1.0f);
      sh[14] = c5 * z * (x * x - y * y);
      sh[15] = c1 * x * (x * x - 3.0f * y * y);
    } else {
      geom[wid][lane][0] = 0.0f;
      geom[wid][lane][1] = 0.0f;
    }
  }
  __syncthreads();

  {  // radial basis: each lane does (edge m = lane&15, half = lane>>4), 64 sines
    int m = lane & 15, half = lane >> 4;
    float u = geom[wid][m][0], pref = geom[wid][m][1];
#pragma unroll 4
    for (int j = 0; j < 64; ++j) {
      int nidx = half * 64 + j + 1;
      rb_lds[wid][m][half * 64 + j] = pref * __sinf(PI_F * (float)nidx * u);
    }
  }
  __syncthreads();

  int ncol  = lane & 15;
  int rbase = (lane < 16) ? 0 : 8;

  // ---- GEMM1: rb[16x128] @ rw1[128x64] -> h[16x64], silu ----
  v16bf a1[4];
#pragma unroll
  for (int kc = 0; kc < 4; ++kc) a1[kc] = make_a_frag(&rb_lds[wid][0][0], NBASIS, kc * 32, lane);
#pragma unroll
  for (int nt = 0; nt < 4; ++nt) {
    v8f acc = {0, 0, 0, 0, 0, 0, 0, 0};
#pragma unroll
    for (int kc = 0; kc < 4; ++kc)
      acc = __builtin_amdgcn_wmma_f32_16x16x32_bf16(false, a1[kc], false,
              bfrag1[(kc * 4 + nt) * 32 + lane], (short)0, acc, false, false);
    int n = nt * 16 + ncol;
    float bias = rb1[n];
#pragma unroll
    for (int j = 0; j < 8; ++j)
      h_lds[wid][rbase + j][n] = siluf(acc[j] + bias);
  }
  __syncthreads();

  v16bf a2[2];
#pragma unroll
  for (int kc = 0; kc < 2; ++kc) a2[kc] = make_a_frag(&h_lds[wid][0][0], HDIM, kc * 32, lane);

  // ---- GEMM2: h[16x64] @ rw2[64x384] -> w_all ----
  for (int nt = 0; nt < 24; ++nt) {
    v8f acc = {0, 0, 0, 0, 0, 0, 0, 0};
#pragma unroll
    for (int kc = 0; kc < 2; ++kc)
      acc = __builtin_amdgcn_wmma_f32_16x16x32_bf16(false, a2[kc], false,
              bfrag2[(kc * 24 + nt) * 32 + lane], (short)0, acc, false, false);
    int col = nt * 16 + ncol;
    float bias = rb2[col];
#pragma unroll
    for (int j = 0; j < 8; ++j) {
      long e = base + rbase + j;
      if (e < N_EDGES) w_all[e * 384 + col] = acc[j] + bias;
    }
  }

  // ---- GEMM3: h[16x64] @ edge_w[64x128] -> ew ----
#pragma unroll
  for (int nt = 0; nt < 8; ++nt) {
    v8f acc = {0, 0, 0, 0, 0, 0, 0, 0};
#pragma unroll
    for (int kc = 0; kc < 2; ++kc)
      acc = __builtin_amdgcn_wmma_f32_16x16x32_bf16(false, a2[kc], false,
              bfragE[(kc * 8 + nt) * 32 + lane], (short)0, acc, false, false);
    int col = nt * 16 + ncol;
    float bias = edge_b[col];
#pragma unroll
    for (int j = 0; j < 8; ++j) {
      long e = base + rbase + j;
      if (e < N_EDGES) ew_all[e * 128 + col] = acc[j] + bias;
    }
  }
}

__global__ void init_f_kernel(const float* __restrict__ ne, const int* __restrict__ species,
                              float* __restrict__ f) {
  int i = blockIdx.x * blockDim.x + threadIdx.x;
  if (i >= N_NODES * 512) return;
  int n = i >> 9, col = i & 511;
  f[i] = (col < 32) ? ne[species[n] * 32 + col] : 0.0f;
}

__global__ void zero_kernel(float* __restrict__ p, int count) {
  int i = blockIdx.x * blockDim.x + threadIdx.x;
  if (i < count) p[i] = 0.0f;
}

__global__ void copy_kernel(const float* __restrict__ src, float* __restrict__ dst, int count) {
  int i = blockIdx.x * blockDim.x + threadIdx.x;
  if (i < count) dst[i] = src[i];
}

// One wave per edge; lane = channel. Scatter messages with hardware float atomics.
__global__ __launch_bounds__(256)
void message_kernel(const int* __restrict__ ei, const float* __restrict__ f,
                    const float* __restrict__ w_all, const float* __restrict__ sh_all,
                    float* __restrict__ agg, int t) {
  int wid = threadIdx.x >> 5, lane = threadIdx.x & 31;
  long e = (long)blockIdx.x * 8 + wid;
  if (e >= N_EDGES) return;
  int s = ei[e], d = ei[N_EDGES + e];
  float sv = f[(long)s * 512 + lane];
  const float* wrow = &w_all[e * 384 + (long)t * 128];
  const float* sh = &sh_all[e * 16];
  if (e + 8 < N_EDGES) __builtin_prefetch(&w_all[(e + 8) * 384 + (long)t * 128], 0, 1);
  float* aggn = &agg[(long)d * 512];
  const int FL[4] = {0, 32, 128, 288};
  const int SO[4] = {0, 1, 4, 9};
  const int DL[4] = {1, 3, 5, 7};
#pragma unroll
  for (int l = 0; l < 4; ++l) {
    float wsv = wrow[l * 32 + lane] * sv;
    float* ap = &aggn[FL[l] + lane * DL[l]];
#pragma unroll
    for (int m = 0; m < 7; ++m)
      if (m < DL[l]) atomAddF(&ap[m], wsv * sh[SO[l] + m]);
  }
}

// One wave per node; lane = output channel d. Weights staged in LDS.
__global__ __launch_bounds__(256)
void node_update_kernel(const float* __restrict__ f, const float* __restrict__ agg,
                        float* __restrict__ fn,
                        const float* __restrict__ sw_g, const float* __restrict__ mw_g,
                        const float* __restrict__ gw_g) {
  __shared__ float sw[4096];
  __shared__ float mw[4096];
  __shared__ float gw[1024];
  __shared__ float n0[8][32];
  for (int i = threadIdx.x; i < 4096; i += 256) { sw[i] = sw_g[i]; mw[i] = mw_g[i]; }
  for (int i = threadIdx.x; i < 1024; i += 256) gw[i] = gw_g[i];
  __syncthreads();

  int wid = threadIdx.x >> 5, d = threadIdx.x & 31;
  long n = (long)blockIdx.x * 8 + wid;  // grid sized exactly: no divergence at barriers
  const float* fr = &f[n * 512];
  const float* ar = &agg[n * 512];

  float acc0 = 0.0f;
  for (int c = 0; c < 32; ++c)
    acc0 += fr[c] * sw[c * 32 + d] + ar[c] * mw[c * 32 + d];
  n0[wid][d] = acc0;
  __syncthreads();

  float gsum = 0.0f;
  for (int c = 0; c < 32; ++c) gsum += n0[wid][c] * gw[c * 32 + d];
  float gate = sigmoidf_(gsum);

  float* fo = &fn[n * 512];
  fo[d] = siluf(acc0);
  const int FL[4] = {0, 32, 128, 288};
  const int DL[4] = {1, 3, 5, 7};
#pragma unroll
  for (int l = 1; l < 4; ++l) {
    const float* swl = &sw[l * 1024];
    const float* mwl = &mw[l * 1024];
    for (int m = 0; m < DL[l]; ++m) {
      float acc = 0.0f;
      for (int c = 0; c < 32; ++c)
        acc += fr[FL[l] + c * DL[l] + m] * swl[c * 32 + d]
             + ar[FL[l] + c * DL[l] + m] * mwl[c * 32 + d];
      fo[FL[l] + d * DL[l] + m] = acc * gate;
    }
  }
}

// One wave per edge: edge_out rows of final output
__global__ __launch_bounds__(256)
void edge_out_kernel(const int* __restrict__ ei, const float* __restrict__ f,
                     const float* __restrict__ ew_all, const float* __restrict__ sh_all,
                     float* __restrict__ out) {
  int wid = threadIdx.x >> 5, lane = threadIdx.x & 31;
  long e = (long)blockIdx.x * 8 + wid;
  if (e >= N_EDGES) return;
  int s = ei[e], d = ei[N_EDGES + e];
  float g = f[(long)s * 512 + lane] + f[(long)d * 512 + lane];
  const float* sh = &sh_all[e * 16];
  const float* ewr = &ew_all[e * 128];
  float* o = &out[((long)N_NODES + e) * 512];
  const int FL[4] = {0, 32, 128, 288};
  const int SO[4] = {0, 1, 4, 9};
  const int DL[4] = {1, 3, 5, 7};
#pragma unroll
  for (int l = 0; l < 4; ++l) {
    float w = ewr[l * 32 + lane] * g;
#pragma unroll
    for (int m = 0; m < 7; ++m)
      if (m < DL[l]) o[FL[l] + lane * DL[l] + m] = w * sh[SO[l] + m];
  }
}

extern "C" void kernel_launch(void* const* d_in, const int* in_sizes, int n_in,
                              void* d_out, int out_size, void* d_ws, size_t ws_size,
                              hipStream_t stream) {
  (void)in_sizes; (void)n_in; (void)out_size; (void)ws_size;
  const float* pos        = (const float*)d_in[0];
  const float* node_embed = (const float*)d_in[1];
  const float* rw1        = (const float*)d_in[2];
  const float* rb1        = (const float*)d_in[3];
  const float* rw2        = (const float*)d_in[4];
  const float* rb2        = (const float*)d_in[5];
  const float* self_w     = (const float*)d_in[6];
  const float* msg_w      = (const float*)d_in[7];
  const float* gate_w     = (const float*)d_in[8];
  const float* edge_w     = (const float*)d_in[9];
  const float* edge_b     = (const float*)d_in[10];
  const int*   species    = (const int*)d_in[11];
  const int*   ei         = (const int*)d_in[12];
  float* out = (float*)d_out;

  char* ws = (char*)d_ws;
  size_t off = 0;
  auto alloc = [&](size_t bytes) { size_t o = off; off = (off + bytes + 255) & ~(size_t)255; return o; };
  size_t o_b1  = alloc(8192 * sizeof(bf16_t));
  size_t o_b2  = alloc(24576 * sizeof(bf16_t));
  size_t o_bE  = alloc(8192 * sizeof(bf16_t));
  size_t o_sh  = alloc((size_t)N_EDGES * 16 * 4);
  size_t o_wal = alloc((size_t)N_EDGES * 384 * 4);
  size_t o_ew  = alloc((size_t)N_EDGES * 128 * 4);
  size_t o_fA  = alloc((size_t)N_NODES * 512 * 4);
  size_t o_fB  = alloc((size_t)N_NODES * 512 * 4);
  size_t o_agg = alloc((size_t)N_NODES * 512 * 4);

  bf16_t* b1 = (bf16_t*)(ws + o_b1);
  bf16_t* b2 = (bf16_t*)(ws + o_b2);
  bf16_t* bE = (bf16_t*)(ws + o_bE);
  float* sh   = (float*)(ws + o_sh);
  float* wall = (float*)(ws + o_wal);
  float* ew   = (float*)(ws + o_ew);
  float* fA   = (float*)(ws + o_fA);
  float* fB   = (float*)(ws + o_fB);
  float* agg  = (float*)(ws + o_agg);

  prep_frag_kernel<<<(8192 + 255) / 256, 256, 0, stream>>>(rw1, b1, 4, 64, 8192);
  prep_frag_kernel<<<(24576 + 255) / 256, 256, 0, stream>>>(rw2, b2, 24, 384, 24576);
  prep_frag_kernel<<<(8192 + 255) / 256, 256, 0, stream>>>(edge_w, bE, 8, 128, 8192);

  edge_mlp_kernel<<<N_EDGES / 64, 128, 0, stream>>>(pos, ei, rb1, rb2, edge_b,
      (const v16bf*)b1, (const v16bf*)b2, (const v16bf*)bE, sh, wall, ew);

  init_f_kernel<<<(N_NODES * 512 + 255) / 256, 256, 0, stream>>>(node_embed, species, fA);

  float* cur = fA;
  float* nxt = fB;
  for (int t = 0; t < 3; ++t) {
    zero_kernel<<<(N_NODES * 512 + 255) / 256, 256, 0, stream>>>(agg, N_NODES * 512);
    message_kernel<<<N_EDGES / 8, 256, 0, stream>>>(ei, cur, wall, sh, agg, t);
    node_update_kernel<<<N_NODES / 8, 256, 0, stream>>>(cur, agg, nxt,
        self_w + (size_t)t * 4096, msg_w + (size_t)t * 4096, gate_w + (size_t)t * 1024);
    float* tmp = cur; cur = nxt; nxt = tmp;
  }

  copy_kernel<<<(N_NODES * 512 + 255) / 256, 256, 0, stream>>>(cur, out, N_NODES * 512);
  edge_out_kernel<<<N_EDGES / 8, 256, 0, stream>>>(ei, cur, ew, sh, out);
}
